// MambaBlock_89232240542341
// MI455X (gfx1250) — matrix operand order repaired and checked by reading, not compiled, not run on hardware
//
#include <hip/hip_runtime.h>
#include <hip/hip_bf16.h>

#define DMODEL   2048
#define DINNER   4096
#define DSTATE   128
#define DCONV    4
#define HEADDIM  64
#define NHEADS   64
#define CONVDIM  (DINNER + 2*DSTATE)            // 4352
#define DINPROJ  (2*DINNER + 2*DSTATE + NHEADS) // 8512
#define BATCH    2
#define SEQ      2048
#define MT       (BATCH*SEQ)                    // 4096
#define EPS_RMS  1e-5f

typedef __bf16 bf16;
typedef __attribute__((ext_vector_type(16))) __bf16 v16bf;
typedef __attribute__((ext_vector_type(8)))  __bf16 v8bf;
typedef __attribute__((ext_vector_type(8)))  float  v8f;

// ---------------------------------------------------------------------------
// C[M,N] = A[M,K] * W[N,K]^T   (fp32 in memory, bf16 WMMA, fp32 accumulate)
// block = 256 threads (8 waves); block tile 128x128; K tile 32.
// wave grid 4(M) x 2(N): each wave owns a 32x64 register tile
// = 2 m-frags x 4 n-frags = 8 WMMA accumulators.
// M must divide 128; N edge handled with clamped loads / guarded stores.
// ---------------------------------------------------------------------------
__global__ __launch_bounds__(256)
void wmma_gemm_nt(const float* __restrict__ A, const float* __restrict__ W,
                  float* __restrict__ C, int M, int N, int K)
{
    __shared__ bf16 As[128][40];  // +8 bf16 pad: 16B-aligned rows, no bank conflicts
    __shared__ bf16 Bs[128][40];

    const int tid  = threadIdx.x;
    const int lane = tid & 31;
    const int wave = tid >> 5;      // 0..7
    const int wm   = wave >> 1;     // 0..3 -> M offset wm*32
    const int wn   = wave & 1;      // 0..1 -> N offset wn*64
    const int half = lane >> 4;     // 0/1
    const int l16  = lane & 15;

    const int m0 = blockIdx.y * 128;
    const int n0 = blockIdx.x * 128;

    // cooperative loader: row = tid>>1 (0..127), 16-float half-row per thread
    const int lr = tid >> 1;
    const int lc = (tid & 1) * 16;
    int wrow = n0 + lr; if (wrow > N - 1) wrow = N - 1;   // clamp N edge

    v8f acc[2][4];
#pragma unroll
    for (int mi = 0; mi < 2; ++mi)
#pragma unroll
        for (int j = 0; j < 4; ++j)
            acc[mi][j] = (v8f){0.f,0.f,0.f,0.f,0.f,0.f,0.f,0.f};

    for (int k0 = 0; k0 < K; k0 += 32) {
        const float* ga = &A[(size_t)(m0 + lr) * K + k0 + lc];
        const float* gb = &W[(size_t)wrow * K + k0 + lc];
        if (k0 + 32 < K) {                       // global_prefetch_b8 on next K tile
            __builtin_prefetch(ga + 32, 0, 1);
            __builtin_prefetch(gb + 32, 0, 1);
        }
#pragma unroll
        for (int v = 0; v < 4; ++v) {
            float4 fa = ((const float4*)ga)[v];
            float4 fb = ((const float4*)gb)[v];
            int c = lc + v * 4;
            // plain casts -> hardware v_cvt f32->bf16 path (fptrunc)
            As[lr][c+0] = (bf16)fa.x; As[lr][c+1] = (bf16)fa.y;
            As[lr][c+2] = (bf16)fa.z; As[lr][c+3] = (bf16)fa.w;
            Bs[lr][c+0] = (bf16)fb.x; Bs[lr][c+1] = (bf16)fb.y;
            Bs[lr][c+2] = (bf16)fb.z; Bs[lr][c+3] = (bf16)fb.w;
        }
        __syncthreads();

        // A fragments (16x32, ISA layout: lane<16 K={0..7,16..23}, lane>=16 K={8..15,24..31})
        v16bf afrag[2];
#pragma unroll
        for (int mi = 0; mi < 2; ++mi) {
            const int ar = wm * 32 + mi * 16 + l16;
            v8bf a_lo = *(const v8bf*)&As[ar][half * 8];
            v8bf a_hi = *(const v8bf*)&As[ar][16 + half * 8];
            afrag[mi] = __builtin_shufflevector(a_lo, a_hi,
                            0,1,2,3,4,5,6,7,8,9,10,11,12,13,14,15);
        }

#pragma unroll
        for (int j = 0; j < 4; ++j) {
            // B fragment (32x16: lane<16 holds K={0..7,16..23} of col n, lane>=16 K={8..15,24..31})
            const int br = wn * 64 + j * 16 + l16;
            v8bf b_lo = *(const v8bf*)&Bs[br][half * 8];
            v8bf b_hi = *(const v8bf*)&Bs[br][16 + half * 8];
            v16bf bfrag = __builtin_shufflevector(b_lo, b_hi,
                            0,1,2,3,4,5,6,7,8,9,10,11,12,13,14,15);
#pragma unroll
            for (int mi = 0; mi < 2; ++mi)
                acc[mi][j] = __builtin_amdgcn_wmma_f32_16x16x32_bf16(
                                 false, afrag[mi], false, bfrag,
                                 (short)0, acc[mi][j], false, false);
        }
        __syncthreads();
    }

    // D layout: VGPR v -> row (half*8 + v), lane l16 -> col
#pragma unroll
    for (int mi = 0; mi < 2; ++mi) {
        const int mrow = m0 + wm * 32 + mi * 16 + half * 8;
#pragma unroll
        for (int j = 0; j < 4; ++j) {
            const int ncol = n0 + wn * 64 + j * 16 + l16;
            if (ncol < N) {
#pragma unroll
                for (int v = 0; v < 8; ++v)
                    C[(size_t)(mrow + v) * N + ncol] = acc[mi][j][v];
            }
        }
    }
}

// ---------------------------------------------------------------------------
// Depthwise causal conv (k=4) over xBC slice of zxbcdt, SiLU, plus softplus(dt)
// ---------------------------------------------------------------------------
__global__ __launch_bounds__(256)
void conv_silu_dt(const float* __restrict__ zxbcdt,
                  const float* __restrict__ conv_w, const float* __restrict__ conv_b,
                  const float* __restrict__ dt_bias,
                  float* __restrict__ xBCc, float* __restrict__ dtbuf)
{
    const int row = blockIdx.x;          // b*SEQ + t
    const int t   = row & (SEQ - 1);
    const int c   = blockIdx.y * 256 + threadIdx.x;
    if (c < CONVDIM) {
        float acc = conv_b[c];
#pragma unroll
        for (int k = 0; k < DCONV; ++k) {
            int tt = t - (DCONV - 1) + k;
            if (tt >= 0)
                acc += zxbcdt[(size_t)(row - (DCONV - 1) + k) * DINPROJ + DINNER + c]
                       * conv_w[c * DCONV + k];
        }
        xBCc[(size_t)row * CONVDIM + c] = acc / (1.f + __expf(-acc));
    }
    if (blockIdx.y == 0 && threadIdx.x < NHEADS) {
        const int h = threadIdx.x;
        float raw = zxbcdt[(size_t)row * DINPROJ + DINNER + CONVDIM + h] + dt_bias[h];
        dtbuf[row * NHEADS + h] = (raw > 20.f) ? raw : __logf(1.f + __expf(raw));
    }
}

// ---------------------------------------------------------------------------
// Selective-state scan. One block per (b, head). State h[64][128] lives in
// registers (32 floats/thread x 256 threads). Next-step B/C/x/dt prefetched
// into registers while current step computes (hides global latency over the
// 2048-step recurrence).
// ---------------------------------------------------------------------------
__global__ __launch_bounds__(256)
void ssm_scan(const float* __restrict__ xBCc, const float* __restrict__ dtbuf,
              const float* __restrict__ A_log, const float* __restrict__ Dskip,
              float* __restrict__ ybuf)
{
    const int blk = blockIdx.x;          // b*NHEADS + h
    const int b   = blk / NHEADS;
    const int hh  = blk % NHEADS;
    const int tid = threadIdx.x;
    const int p   = tid >> 2;            // 0..63 head-dim row
    const int q   = tid & 3;             // n-chunk
    const int nb  = q * 32;

    __shared__ float Bs[DSTATE];
    __shared__ float Cs[DSTATE];
    __shared__ float xs[HEADDIM];
    __shared__ float red[256];

    const float Ah = -__expf(A_log[hh]);
    const float Dh = Dskip[hh];

    float h[32];
#pragma unroll
    for (int i = 0; i < 32; ++i) h[i] = 0.f;

    // prefetch t = 0
    float rB = 0.f, rC = 0.f, rX = 0.f, rDT = 0.f;
    {
        const size_t row = (size_t)b * SEQ;
        if (tid < DSTATE) rB = xBCc[row * CONVDIM + DINNER + tid];
        else              rC = xBCc[row * CONVDIM + DINNER + DSTATE + (tid - DSTATE)];
        if (tid < HEADDIM) rX = xBCc[row * CONVDIM + hh * HEADDIM + tid];
        rDT = dtbuf[row * NHEADS + hh];
    }

    for (int t = 0; t < SEQ; ++t) {
        if (tid < DSTATE) Bs[tid] = rB;
        else              Cs[tid - DSTATE] = rC;
        if (tid < HEADDIM) xs[tid] = rX;
        const float dtv = rDT;
        __syncthreads();

        if (t + 1 < SEQ) {               // issue next-step loads early
            const size_t row = (size_t)b * SEQ + t + 1;
            if (tid < DSTATE) rB = xBCc[row * CONVDIM + DINNER + tid];
            else              rC = xBCc[row * CONVDIM + DINNER + DSTATE + (tid - DSTATE)];
            if (tid < HEADDIM) rX = xBCc[row * CONVDIM + hh * HEADDIM + tid];
            rDT = dtbuf[row * NHEADS + hh];
        }

        const float dA  = __expf(dtv * Ah);
        const float dtx = dtv * xs[p];
        float partial = 0.f;
#pragma unroll
        for (int i = 0; i < 32; ++i) {
            h[i] = h[i] * dA + dtx * Bs[nb + i];
            partial += h[i] * Cs[nb + i];
        }
        red[tid] = partial;
        __syncthreads();
        if (q == 0) {
            const size_t row = (size_t)b * SEQ + t;
            float y = red[tid] + red[tid + 1] + red[tid + 2] + red[tid + 3] + Dh * xs[p];
            ybuf[(row * NHEADS + hh) * HEADDIM + p] = y;
        }
        __syncthreads();
    }
}

// ---------------------------------------------------------------------------
// y = y * silu(z); RMS-normalize over D_INNER; scale by norm_w. In-place on ybuf.
// ---------------------------------------------------------------------------
__global__ __launch_bounds__(256)
void gated_rmsnorm(const float* __restrict__ zxbcdt, const float* __restrict__ norm_w,
                   float* __restrict__ ybuf)
{
    const int row = blockIdx.x;
    const int tid = threadIdx.x;
    __shared__ float red[256];

    float g[16];
    float ss = 0.f;
#pragma unroll
    for (int i = 0; i < 16; ++i) {
        const int e = tid + i * 256;
        float y = ybuf[(size_t)row * DINNER + e];
        float z = zxbcdt[(size_t)row * DINPROJ + e];
        float v = y * (z / (1.f + __expf(-z)));
        g[i] = v;
        ss += v * v;
    }
    red[tid] = ss;
    __syncthreads();
    for (int s = 128; s > 0; s >>= 1) {
        if (tid < s) red[tid] += red[tid + s];
        __syncthreads();
    }
    const float scale = rsqrtf(red[0] / (float)DINNER + EPS_RMS);
#pragma unroll
    for (int i = 0; i < 16; ++i) {
        const int e = tid + i * 256;
        ybuf[(size_t)row * DINNER + e] = g[i] * scale * norm_w[e];
    }
}

// ---------------------------------------------------------------------------
extern "C" void kernel_launch(void* const* d_in, const int* in_sizes, int n_in,
                              void* d_out, int out_size, void* d_ws, size_t ws_size,
                              hipStream_t stream)
{
    const float* x       = (const float*)d_in[0];
    // d_in[1]=cos, d_in[2]=sin (unused by reference)
    const float* in_w    = (const float*)d_in[3];   // (DINPROJ, DMODEL)
    const float* conv_w  = (const float*)d_in[4];   // (CONVDIM, DCONV)
    const float* conv_b  = (const float*)d_in[5];
    const float* dt_bias = (const float*)d_in[6];
    const float* A_log   = (const float*)d_in[7];
    const float* Dskip   = (const float*)d_in[8];
    const float* norm_w  = (const float*)d_in[9];
    const float* out_w   = (const float*)d_in[10];  // (DMODEL, DINNER)
    float* out = (float*)d_out;

    char* ws = (char*)d_ws;
    float* zxbcdt = (float*)ws;  ws += (size_t)MT * DINPROJ * sizeof(float);
    float* xBCc   = (float*)ws;  ws += (size_t)MT * CONVDIM * sizeof(float);
    float* dtbuf  = (float*)ws;  ws += (size_t)MT * NHEADS  * sizeof(float);
    float* ybuf   = (float*)ws;

    // 1) in_proj: zxbcdt[MT, DINPROJ] = x[MT, DMODEL] * in_w^T
    wmma_gemm_nt<<<dim3((DINPROJ + 127) / 128, MT / 128), 256, 0, stream>>>(
        x, in_w, zxbcdt, MT, DINPROJ, DMODEL);

    // 2) causal depthwise conv + SiLU + softplus(dt)
    conv_silu_dt<<<dim3(MT, CONVDIM / 256), 256, 0, stream>>>(
        zxbcdt, conv_w, conv_b, dt_bias, xBCc, dtbuf);

    // 3) sequential SSM scan, one block per (batch, head)
    ssm_scan<<<dim3(BATCH * NHEADS), 256, 0, stream>>>(
        xBCc, dtbuf, A_log, Dskip, ybuf);

    // 4) gated RMSNorm (in place on ybuf)
    gated_rmsnorm<<<dim3(MT), 256, 0, stream>>>(zxbcdt, norm_w, ybuf);

    // 5) out_proj: out[MT, DMODEL] = ybuf[MT, DINNER] * out_w^T
    wmma_gemm_nt<<<dim3(DMODEL / 128, MT / 128), 256, 0, stream>>>(
        ybuf, out_w, out, MT, DMODEL, DINNER);
}
